// Classifier_67602785239058
// MI455X (gfx1250) — compile-verified, matrix operand order checked by reading, not run on hardware
//
#include <hip/hip_runtime.h>
#include <math.h>

typedef __attribute__((ext_vector_type(16))) _Float16 v16h;
typedef __attribute__((ext_vector_type(8)))  _Float16 v8h;
typedef __attribute__((ext_vector_type(8)))  float    v8f;

#define IN_DIM 128
#define HID    64

// Deterministic per-element dropout hash (p = 0.5, keep -> x2).
__device__ __forceinline__ bool keep_drop(unsigned idx, unsigned seed) {
  unsigned x = idx * 0x9E3779B9u ^ seed;
  x ^= x >> 16; x *= 0x7feb352du;
  x ^= x >> 15; x *= 0x846ca68bu;
  x ^= x >> 16;
  return (x & 1u) != 0u;
}

// deg = 1 (self loop), S = 0
__global__ void init_k(float* deg, float* S, int n) {
  int i = blockIdx.x * blockDim.x + threadIdx.x;
  if (i < n)  deg[i] = 1.0f;
  if (i < 64) S[i]   = 0.0f;
}

__global__ void deg_k(const int* __restrict__ dst, float* deg, int E) {
  int e = blockIdx.x * blockDim.x + threadIdx.x;
  if (e < E) atomicAdd(&deg[dst[e]], 1.0f);
}

__global__ void norm_k(float* deg, int n) {
  int i = blockIdx.x * blockDim.x + threadIdx.x;
  if (i < n) deg[i] = rsqrtf(deg[i]); // deg >= 1 always (self loop)
}

// Transpose weights to N-major fp16 so WMMA B-fragments load contiguously.
__global__ void prep_w_k(const float* __restrict__ w1, const float* __restrict__ w2,
                         _Float16* __restrict__ w1t, _Float16* __restrict__ w2t) {
  int i = blockIdx.x * blockDim.x + threadIdx.x;
  if (i < IN_DIM * HID) { int k = i / HID, nn = i % HID; w1t[nn * IN_DIM + k] = (_Float16)w1[i]; }
  if (i < HID * HID)    { int k = i / HID, nn = i % HID; w2t[nn * HID    + k] = (_Float16)w2[i]; }
}

// out[M x 64] = (A[M x K] * pre-scale) @ W[K x 64]  (W given N-major fp16)
// PRE:  scale each A row by nrm[row] (fused into the f32->f16 convert)
// EPI:  out = dropout(relu(acc * nrm[row]))
// DUAL: also store the raw tile to out2 (self-loop init of the aggregation buffer)
template<int K, bool PRE, bool EPI, bool DUAL>
__global__ __launch_bounds__(256) void gemm_wmma_k(
    const float* __restrict__ A, const _Float16* __restrict__ Wt,
    const float* __restrict__ nrm, float* __restrict__ out,
    float* __restrict__ out2, int Mtiles, unsigned seed) {
  __shared__ _Float16 sW[HID * K];
  for (int i = threadIdx.x; i < HID * K / 8; i += 256)
    ((v8h*)sW)[i] = ((const v8h*)Wt)[i];        // global b128 -> ds b128
  __syncthreads();

  const int wave    = threadIdx.x >> 5;
  const int lane    = threadIdx.x & 31;
  const int tile    = blockIdx.x * 8 + wave;
  if (tile >= Mtiles) return;

  const int halfSel = lane >> 4;      // 0: lanes 0-15, 1: lanes 16-31
  const int mrow    = tile * 16 + (lane & 15);
  const int koff    = halfSel * 8;

  float scale = 1.0f;
  if (PRE) scale = nrm[mrow];

  v8f acc[4];
#pragma unroll
  for (int t = 0; t < 4; ++t)
#pragma unroll
    for (int i = 0; i < 8; ++i) acc[t][i] = 0.0f;

#pragma unroll
  for (int ks = 0; ks < K / 32; ++ks) {
    const int k0 = ks * 32;
    // A fragment per ISA 16-bit A layout: lane row = M, halves {k0+koff..+7} and {k0+koff+16..+23}
    const float* ap = A + (size_t)mrow * K + k0 + koff;
    float4 a0 = *(const float4*)(ap +  0);
    float4 a1 = *(const float4*)(ap +  4);
    float4 a2 = *(const float4*)(ap + 16);
    float4 a3 = *(const float4*)(ap + 20);
    v16h af;
    af[0]  = (_Float16)(a0.x * scale); af[1]  = (_Float16)(a0.y * scale);
    af[2]  = (_Float16)(a0.z * scale); af[3]  = (_Float16)(a0.w * scale);
    af[4]  = (_Float16)(a1.x * scale); af[5]  = (_Float16)(a1.y * scale);
    af[6]  = (_Float16)(a1.z * scale); af[7]  = (_Float16)(a1.w * scale);
    af[8]  = (_Float16)(a2.x * scale); af[9]  = (_Float16)(a2.y * scale);
    af[10] = (_Float16)(a2.z * scale); af[11] = (_Float16)(a2.w * scale);
    af[12] = (_Float16)(a3.x * scale); af[13] = (_Float16)(a3.y * scale);
    af[14] = (_Float16)(a3.z * scale); af[15] = (_Float16)(a3.w * scale);

#pragma unroll
    for (int nt = 0; nt < 4; ++nt) {
      const int ncol = nt * 16 + (lane & 15);
      const v8h* bp = (const v8h*)(&sW[ncol * K + k0 + koff]);
      v8h blo = bp[0];
      v8h bhi = bp[2];
      v16h bf;
#pragma unroll
      for (int i = 0; i < 8; ++i) { bf[i] = blo[i]; bf[8 + i] = bhi[i]; }
      acc[nt] = __builtin_amdgcn_wmma_f32_16x16x32_f16(
          false, af, false, bf, (short)0, acc[nt], false, false);
    }
  }

  // D layout: VGPR v -> row tile*16 + v + 8*halfSel, col = nt*16 + (lane&15)
#pragma unroll
  for (int nt = 0; nt < 4; ++nt) {
#pragma unroll
    for (int v = 0; v < 8; ++v) {
      const int r = tile * 16 + v + halfSel * 8;
      const int c = nt * 16 + (lane & 15);
      float val = acc[nt][v];
      if (EPI) {
        float nm = nrm[r];
        val = fmaxf(val * nm, 0.0f);
        val = keep_drop((unsigned)(r * HID + c), seed) ? val * 2.0f : 0.0f;
      }
      out[(size_t)r * HID + c] = val;
      if (DUAL) out2[(size_t)r * HID + c] = val;   // self-loop init, saves a copy pass
    }
  }
}

// One wave per edge: agg[dst] += h[src], 2 contiguous floats per lane (coalesced).
// Destination row prefetched toward GL2 ahead of the atomic RMW stream.
__global__ __launch_bounds__(256) void scatter_k(
    const float* __restrict__ h, float* agg,
    const int* __restrict__ src, const int* __restrict__ dst, int E) {
  int gt = blockIdx.x * blockDim.x + threadIdx.x;
  int e = gt >> 5, lane = gt & 31;
  if (e >= E) return;
  int s = src[e], d = dst[e];
  float* dp = agg + (size_t)d * HID + lane * 2;
  __builtin_prefetch(dp, 1, 1);                     // global_prefetch_b8
  const float2 v = *(const float2*)(h + (size_t)s * HID + lane * 2);
  atomicAdd(dp + 0, v.x);
  atomicAdd(dp + 1, v.y);
}

// Layer-1 epilogue fused with layer-2 pre-scale: H = agg2init = dropout(relu(agg*nm))*nm
__global__ void post1_k(const float* agg, const float* __restrict__ nrm,
                        float* H, float* agg2, int total, unsigned seed) {
  int i = blockIdx.x * blockDim.x + threadIdx.x;
  if (i >= total) return;
  float nm = nrm[i >> 6];
  float v = fmaxf(agg[i] * nm, 0.0f);
  v = keep_drop((unsigned)i, seed) ? v * 2.0f : 0.0f;
  v *= nm;                      // layer-2's leading h*norm
  H[i] = v;
  agg2[i] = v;                  // self-loop init for layer-2 aggregation
}

// Column sums of H2 (mean-of-conv shortcut: conv is linear, so mean(conv(x)) = conv(colsum)/N + b).
__global__ void colsum_k(const float* __restrict__ h, float* S, int n) {
  int tid = blockIdx.x * blockDim.x + threadIdx.x;
  int c = tid & 63;
  int r0 = tid >> 6;
  int stride = (gridDim.x * blockDim.x) >> 6;
  float s = 0.0f;
  for (int r = r0; r < n; r += stride) s += h[(size_t)r * HID + c];
  atomicAdd(&S[c], s);
}

// o[e] = conv(S)[e]/N + b[c];  also emits d_out[n] = ref_radius
__global__ void mean_k(const float* __restrict__ S, const float* __restrict__ cw,
                       const float* __restrict__ cb, float* o, int n,
                       const float* __restrict__ ref, float* dout) {
  int e = threadIdx.x;
  if (e < 124) {
    int c = e / 62, j = e % 62;
    float s = cw[c * 3 + 0] * S[j] + cw[c * 3 + 1] * S[j + 1] + cw[c * 3 + 2] * S[j + 2];
    o[e] = s / (float)n + cb[c];
  }
  if (e == 124) dout[n] = ref[0];
}

// One wave per node: recompute 3-tap conv, accumulate (h-o+1e-6)^2, wave32 shuffle reduce.
__global__ __launch_bounds__(256) void radius_k(
    const float* __restrict__ h, const float* __restrict__ o,
    const float* __restrict__ cw, const float* __restrict__ cb,
    const float* __restrict__ ref, float* __restrict__ dout, int n) {
  int gt = blockIdx.x * blockDim.x + threadIdx.x;
  int node = gt >> 5, lane = gt & 31;
  if (node >= n) return;
  const float* x = h + (size_t)node * HID;
  float w00 = cw[0], w01 = cw[1], w02 = cw[2];
  float w10 = cw[3], w11 = cw[4], w12 = cw[5];
  float b0 = cb[0], b1 = cb[1];
  float s = 0.0f;
  for (int e = lane; e < 124; e += 32) {
    int c = (e >= 62) ? 1 : 0;
    int j = e - c * 62;
    float v = c ? (w10 * x[j] + w11 * x[j + 1] + w12 * x[j + 2] + b1)
                : (w00 * x[j] + w01 * x[j + 1] + w02 * x[j + 2] + b0);
    float d = v - o[e] + 1e-6f;
    s += d * d;
  }
#pragma unroll
  for (int m = 16; m >= 1; m >>= 1) s += __shfl_xor(s, m, 32);
  if (lane == 0) {
    float r = sqrtf(s) - ref[0];
    r = fminf(fmaxf(r, 1e-4f), 1.0f - 1e-4f);
    dout[node] = r;
  }
}

extern "C" void kernel_launch(void* const* d_in, const int* in_sizes, int n_in,
                              void* d_out, int out_size, void* d_ws, size_t ws_size,
                              hipStream_t stream) {
  const float* feat = (const float*)d_in[0];
  const float* w1   = (const float*)d_in[1];
  const float* w2   = (const float*)d_in[2];
  const float* cw   = (const float*)d_in[3];
  const float* cb   = (const float*)d_in[4];
  const float* ref  = (const float*)d_in[5];
  const int*   src  = (const int*)d_in[6];
  const int*   dst  = (const int*)d_in[7];
  const int n = in_sizes[0] / IN_DIM;   // 100000 (multiple of 16)
  const int E = in_sizes[6];            // 1600000
  float* out = (float*)d_out;

  // Workspace carving (~52 MB total, 256B aligned slices)
  char* ws = (char*)d_ws;
  size_t off = 0;
  auto take = [&](size_t bytes) -> char* {
    char* p = ws + off;
    off = (off + bytes + 255) & ~(size_t)255;
    return p;
  };
  float*    nrm  = (float*)take((size_t)n * 4);          // deg then norm (in place)
  _Float16* w1t  = (_Float16*)take((size_t)IN_DIM * HID * 2);
  _Float16* w2t  = (_Float16*)take((size_t)HID * HID * 2);
  float*    S    = (float*)take(64 * 4);
  float*    om   = (float*)take(124 * 4);
  float*    bufA = (float*)take((size_t)n * HID * 4);    // H1 / Hs / H2
  float*    bufB = (float*)take((size_t)n * HID * 4);    // agg1 / agg2
  (void)ws_size; (void)n_in; (void)out_size;

  const int Mtiles = (n + 15) / 16;
  const int gblk   = (Mtiles + 7) / 8;
  const unsigned edgeBlocks = (unsigned)(((size_t)E * 32 + 255) / 256);

  init_k<<<(n + 255) / 256, 256, 0, stream>>>(nrm, S, n);
  deg_k<<<(E + 255) / 256, 256, 0, stream>>>(dst, nrm, E);
  norm_k<<<(n + 255) / 256, 256, 0, stream>>>(nrm, n);
  prep_w_k<<<(IN_DIM * HID + 255) / 256, 256, 0, stream>>>(w1, w2, w1t, w2t);

  // Layer 1: H1 = (feat*norm) @ w1 (WMMA, dual-store: bufA = H1, bufB = H1 self-loop init)
  gemm_wmma_k<IN_DIM, true, false, true><<<gblk, 256, 0, stream>>>(
      feat, w1t, nrm, bufA, bufB, Mtiles, 0u);
  scatter_k<<<edgeBlocks, 256, 0, stream>>>(bufA, bufB, src, dst, E);
  post1_k<<<(n * HID + 255) / 256, 256, 0, stream>>>(bufB, nrm, bufA, bufB, n * HID, 0x13579BDFu);

  // Layer 2: agg2 = Hs + scatter(edges); H2 = dropout(relu((agg2 @ w2) * norm))
  scatter_k<<<edgeBlocks, 256, 0, stream>>>(bufA, bufB, src, dst, E);
  gemm_wmma_k<HID, false, true, false><<<gblk, 256, 0, stream>>>(
      bufB, w2t, nrm, bufA, nullptr, Mtiles, 0x2468ACE1u);

  // Mean of conv via column sums (conv is linear), then per-node radius.
  colsum_k<<<512, 256, 0, stream>>>(bufA, S, n);
  mean_k<<<1, 128, 0, stream>>>(S, cw, cb, om, n, ref, out);
  radius_k<<<(unsigned)(((size_t)n * 32 + 255) / 256), 256, 0, stream>>>(bufA, om, cw, cb, ref, out, n);
}
